// Block_9380208574648
// MI455X (gfx1250) — matrix operand order repaired and checked by reading, not compiled, not run on hardware
//
#include <hip/hip_runtime.h>
#include <hip/hip_bf16.h>
#include <math.h>

// ---------------- problem constants (match reference) ----------------
constexpr int Bb   = 4;      // batch
constexpr int Ll   = 2048;   // sequence length
constexpr int Dd   = 512;    // model dim
constexpr int Ee   = 1024;   // expanded dim
constexpr int Nst  = 16;     // SSM state dim
constexpr int Kc   = 4;      // conv kernel
constexpr int Rr   = 32;     // dt rank
constexpr int ROWS = Bb * Ll;        // 8192 token rows
constexpr int DBLC = Rr + 2 * Nst;   // 64

typedef __attribute__((ext_vector_type(16))) _Float16 v16h;
typedef __attribute__((ext_vector_type(8)))  _Float16 v8h;
typedef __attribute__((ext_vector_type(8)))  float    v8f;
typedef unsigned int u32x4 __attribute__((ext_vector_type(4)));
typedef int          i32x8 __attribute__((ext_vector_type(8)));
typedef int          i32x4 __attribute__((ext_vector_type(4)));

#define USE_TDM (__has_builtin(__builtin_amdgcn_tensor_load_to_lds) && \
                 __has_builtin(__builtin_amdgcn_s_wait_tensorcnt))

// =====================================================================
// TDM 2-D tile load: global (row-major, rowStride elems of 2B) -> LDS
// tile = tileY rows x tileX halves, appended contiguously in LDS.
// Descriptor packing per CDNA5 ISA ch.8 (D# groups 0/1; groups 2/3 zero).
// This toolchain exposes the 6-arg builtin:
//   (u32x4 g0, i32x8 g1, i32x4 g2, i32x4 g3, i32x8 pad, i32 cpol)
// =====================================================================
#if USE_TDM
__device__ __forceinline__ void tdm_load_tile_f16(const _Float16* gptr,
                                                  unsigned lds_byte_off,
                                                  unsigned tileX, unsigned tileY,
                                                  unsigned tensorX, unsigned tensorY,
                                                  unsigned long long rowStride)
{
    unsigned long long ga = (unsigned long long)(uintptr_t)gptr;
    u32x4 g0;
    g0[0] = 1u;                                            // count=1, no gather
    g0[1] = lds_byte_off;                                  // lds_addr
    g0[2] = (unsigned)ga;                                  // global_addr[31:0]
    g0[3] = (unsigned)((ga >> 32) & 0x01FFFFFFu) | (2u << 30); // addr[56:32] | type=2
    i32x8 g1;
    g1[0] = (int)(1u << 16);                               // data_size=1 (2 bytes)
    g1[1] = (int)((tensorX & 0xFFFFu) << 16);              // tensor_dim0[15:0]
    g1[2] = (int)(((tensorX >> 16) & 0xFFFFu) |
                  ((tensorY & 0xFFFFu) << 16));            // tensor_dim0[31:16]|dim1[15:0]
    g1[3] = (int)(((tensorY >> 16) & 0xFFFFu) |
                  (tileX << 16));                          // tensor_dim1[31:16]|tile_dim0
    g1[4] = (int)(tileY & 0xFFFFu);                        // tile_dim1 (tile_dim2=0)
    g1[5] = (int)(unsigned)(rowStride & 0xFFFFFFFFull);    // tensor_dim0_stride[31:0]
    g1[6] = (int)(unsigned)((rowStride >> 32) & 0xFFFFull);// stride[47:32]
    g1[7] = 0;
    i32x4 g2 = {0, 0, 0, 0};
    i32x4 g3 = {0, 0, 0, 0};
    i32x8 g4 = {0, 0, 0, 0, 0, 0, 0, 0};
    __builtin_amdgcn_tensor_load_to_lds(g0, g1, g2, g3, g4, 0);
}
#endif

// =====================================================================
// RMSNorm: x (ROWS x D) f32 -> nrm16 (ROWS x D) f16
// =====================================================================
__global__ void k_rmsnorm(const float* __restrict__ x,
                          const float* __restrict__ w,
                          _Float16* __restrict__ o)
{
    const int row = blockIdx.x;
    const int t   = threadIdx.x;          // 0..127
    const float* xr = x + (size_t)row * Dd;
    float4 v = ((const float4*)xr)[t];
    float ss = v.x * v.x + v.y * v.y + v.z * v.z + v.w * v.w;
    #pragma unroll
    for (int m = 16; m; m >>= 1) ss += __shfl_xor(ss, m, 32);
    __shared__ float red[4];
    if ((t & 31) == 0) red[t >> 5] = ss;
    __syncthreads();
    float tot = red[0] + red[1] + red[2] + red[3];
    float r = rsqrtf(tot * (1.0f / Dd) + 1e-6f);
    _Float16* orow = o + (size_t)row * Dd;
    const int d0 = t * 4;
    orow[d0 + 0] = (_Float16)(v.x * r * w[d0 + 0]);
    orow[d0 + 1] = (_Float16)(v.y * r * w[d0 + 1]);
    orow[d0 + 2] = (_Float16)(v.z * r * w[d0 + 2]);
    orow[d0 + 3] = (_Float16)(v.w * r * w[d0 + 3]);
}

// =====================================================================
// transpose + f32->f16: src (Kdim x Ndim) -> dst (Ndim x Kdim)
// 32x32 tile via LDS, block = 256 (32x8)
// =====================================================================
__global__ void k_cvt_transpose(const float* __restrict__ s,
                                _Float16* __restrict__ d,
                                int Kdim, int Ndim)
{
    __shared__ float tile[32][33];
    const int kb = blockIdx.x * 32, nb = blockIdx.y * 32;
    const int tx = threadIdx.x & 31, ty = threadIdx.x >> 5;
    #pragma unroll
    for (int i = ty; i < 32; i += 8)
        tile[i][tx] = s[(size_t)(kb + i) * Ndim + nb + tx];
    __syncthreads();
    #pragma unroll
    for (int i = ty; i < 32; i += 8)
        d[(size_t)(nb + i) * Kdim + kb + tx] = (_Float16)tile[tx][i];
}

// =====================================================================
// WMMA GEMM:  C[M,N] f32 = A16[M,K] * Bt16[N,K]^T  (+ residual)
// A row-major MxK; B pre-transposed NxK.  M%128==0, N%64==0, K%32==0.
// Block 256 threads (8 waves); block tile 128x64; wave tile 16x64.
// TDM double-buffered staging when available, else vectorized loads.
// =====================================================================
__global__ void k_gemm_wmma(const _Float16* __restrict__ A16,
                            const _Float16* __restrict__ Bt16,
                            float* __restrict__ C,
                            const float* __restrict__ resid,   // nullable
                            int M, int N, int K)
{
    __shared__ __align__(32) _Float16 As[2][128 * 32];   // 2 x 8 KB
    __shared__ __align__(32) _Float16 Bs[2][64 * 32];    // 2 x 4 KB

    const int t    = threadIdx.x;
    const int wid  = t >> 5;
    const int lane = t & 31;
    const int lh   = lane & 15;
    const int lg   = lane >> 4;

    const int n0 = blockIdx.x * 64;
    const int m0 = blockIdx.y * 128;
    const int nk = K >> 5;

    v8f acc0 = {}, acc1 = {}, acc2 = {}, acc3 = {};

#if USE_TDM
    // ---------------- TDM double-buffered pipeline ----------------
    const unsigned ldsA0 = (unsigned)(uintptr_t)&As[0][0];
    const unsigned ldsA1 = (unsigned)(uintptr_t)&As[1][0];
    const unsigned ldsB0 = (unsigned)(uintptr_t)&Bs[0][0];
    const unsigned ldsB1 = (unsigned)(uintptr_t)&Bs[1][0];

    if (t < 32) {   // wave 0 drives the tensor data mover
        tdm_load_tile_f16(A16 + (size_t)m0 * K, ldsA0, 32, 128,
                          (unsigned)K, (unsigned)M, (unsigned long long)K);
        tdm_load_tile_f16(Bt16 + (size_t)n0 * K, ldsB0, 32, 64,
                          (unsigned)K, (unsigned)N, (unsigned long long)K);
        __builtin_amdgcn_s_wait_tensorcnt(0);
    }
    __syncthreads();

    for (int ks = 0; ks < nk; ++ks) {
        const int cur = ks & 1;
        if (t < 32 && (ks + 1) < nk) {   // prefetch next tiles into other buffer
            const int k1 = (ks + 1) << 5;
            tdm_load_tile_f16(A16 + (size_t)m0 * K + k1, (ks & 1) ? ldsA0 : ldsA1,
                              32, 128, (unsigned)K, (unsigned)M, (unsigned long long)K);
            tdm_load_tile_f16(Bt16 + (size_t)n0 * K + k1, (ks & 1) ? ldsB0 : ldsB1,
                              32, 64, (unsigned)K, (unsigned)N, (unsigned long long)K);
        }
        // ---- compute from current buffer ----
        {
            const _Float16* ap = &As[cur][(wid * 16 + lh) * 32 + lg * 8];
            v8h a0 = *(const v8h*)ap;
            v8h a1 = *(const v8h*)(ap + 16);
            v16h a = __builtin_shufflevector(a0, a1, 0, 1, 2, 3, 4, 5, 6, 7,
                                                     8, 9, 10, 11, 12, 13, 14, 15);
            #pragma unroll
            for (int c = 0; c < 4; ++c) {
                const _Float16* bp = &Bs[cur][(c * 16 + lh) * 32 + lg * 8];
                const v16h b = __builtin_shufflevector(*(const v8h*)bp, *(const v8h*)(bp + 16), 0, 1, 2, 3, 4, 5, 6, 7, 8, 9, 10, 11, 12, 13, 14, 15);
                v8f* accp = (c == 0) ? &acc0 : (c == 1) ? &acc1 : (c == 2) ? &acc2 : &acc3;
                *accp = __builtin_amdgcn_wmma_f32_16x16x32_f16(
                            false, a, false, b, (short)0, *accp, false, false);
            }
        }
        if (t < 32 && (ks + 1) < nk) __builtin_amdgcn_s_wait_tensorcnt(0);
        __syncthreads();
    }
#else
    // ---------------- fallback: vectorized load staging ----------------
    for (int ks = 0; ks < nk; ++ks) {
        const int k0 = ks << 5;
        {   // A: 128x32 halves, 16 halves / thread
            const int r = t >> 1, cseg = (t & 1) * 16;
            const _Float16* g = A16 + (size_t)(m0 + r) * K + k0 + cseg;
            *(v8h*)&As[0][r * 32 + cseg]     = *(const v8h*)g;
            *(v8h*)&As[0][r * 32 + cseg + 8] = *(const v8h*)(g + 8);
        }
        {   // Bt: 64x32 halves, 8 halves / thread (already transposed in global)
            const int r = t >> 2, seg = (t & 3) * 8;
            *(v8h*)&Bs[0][r * 32 + seg] =
                *(const v8h*)(Bt16 + (size_t)(n0 + r) * K + k0 + seg);
        }
        __syncthreads();
        const _Float16* ap = &As[0][(wid * 16 + lh) * 32 + lg * 8];
        v8h a0 = *(const v8h*)ap;
        v8h a1 = *(const v8h*)(ap + 16);
        v16h a = __builtin_shufflevector(a0, a1, 0, 1, 2, 3, 4, 5, 6, 7,
                                                 8, 9, 10, 11, 12, 13, 14, 15);
        #pragma unroll
        for (int c = 0; c < 4; ++c) {
            const _Float16* bp = &Bs[0][(c * 16 + lh) * 32 + lg * 8];
            const v16h b = __builtin_shufflevector(*(const v8h*)bp, *(const v8h*)(bp + 16), 0, 1, 2, 3, 4, 5, 6, 7, 8, 9, 10, 11, 12, 13, 14, 15);
            v8f* accp = (c == 0) ? &acc0 : (c == 1) ? &acc1 : (c == 2) ? &acc2 : &acc3;
            *accp = __builtin_amdgcn_wmma_f32_16x16x32_f16(
                        false, a, false, b, (short)0, *accp, false, false);
        }
        __syncthreads();
    }
#endif

    // ---- epilogue ----
    v8f accs[4] = {acc0, acc1, acc2, acc3};
    #pragma unroll
    for (int c = 0; c < 4; ++c) {
        #pragma unroll
        for (int r = 0; r < 8; ++r) {
            const int row = m0 + wid * 16 + lg * 8 + r;
            const int col = n0 + c * 16 + lh;
            float v = accs[c][r];
            if (resid) v += resid[(size_t)row * N + col];
            C[(size_t)row * N + col] = v;
        }
    }
}

// =====================================================================
// Depthwise causal conv (K=4) + SiLU over the xs half of xz.
// =====================================================================
__global__ void k_conv_silu(const float* __restrict__ xz,
                            const float* __restrict__ cw,
                            const float* __restrict__ cb,
                            float* __restrict__ xs,
                            _Float16* __restrict__ xs16)
{
    const int idx = blockIdx.x * blockDim.x + threadIdx.x;   // over B*L*E
    if (idx >= Bb * Ll * Ee) return;
    const int e = idx % Ee;
    const int l = (idx / Ee) % Ll;
    const int b = idx / (Ee * Ll);
    float acc = cb[e];
    #pragma unroll
    for (int k = 0; k < Kc; ++k) {
        const int ls = l - (Kc - 1) + k;
        if (ls >= 0)
            acc = fmaf(xz[((size_t)(b * Ll + ls)) * (2 * Ee) + e], cw[e * Kc + k], acc);
    }
    const float s = acc / (1.0f + __expf(-acc));   // silu
    xs[idx]   = s;
    xs16[idx] = (_Float16)s;
}

// =====================================================================
// dtin16 (ROWS x 32) = f16(dbl[:, :32])
// =====================================================================
__global__ void k_dtprep(const float* __restrict__ dbl, _Float16* __restrict__ dtin16)
{
    const int i = blockIdx.x * blockDim.x + threadIdx.x;
    if (i >= ROWS * Rr) return;
    const int r = i / Rr, c = i % Rr;
    dtin16[i] = (_Float16)dbl[(size_t)r * DBLC + c];
}

// =====================================================================
// dt = softplus(dtraw + b_dt)  (in place)
// =====================================================================
__global__ void k_softplus(float* __restrict__ dt, const float* __restrict__ b_dt)
{
    const int i = blockIdx.x * blockDim.x + threadIdx.x;
    if (i >= ROWS * Ee) return;
    const float x = dt[i] + b_dt[i % Ee];
    dt[i] = (x > 20.0f) ? x : log1pf(__expf(x));
}

// =====================================================================
// Fused selective scan (wave = 2 channels x 16 states) + gate -> y16
// =====================================================================
__global__ void k_scan(const float* __restrict__ dt,
                       const float* __restrict__ xs,
                       const float* __restrict__ xz,
                       const float* __restrict__ dbl,
                       const float* __restrict__ A_log,
                       const float* __restrict__ D_skip,
                       _Float16* __restrict__ y16)
{
    const int wglobal = (blockIdx.x * blockDim.x + threadIdx.x) >> 5;
    const int lane = threadIdx.x & 31;
    const int n  = lane & 15;
    const int lg = lane >> 4;
    const int b  = wglobal / (Ee / 2);
    const int ep = wglobal % (Ee / 2);
    const int e  = ep * 2 + lg;

    const float An = -__expf(A_log[e * Nst + n]);
    const float Dk = D_skip[e];
    float h = 0.0f;
    const size_t rowbase = (size_t)b * Ll;

    for (int l = 0; l < Ll; ++l) {
        const size_t re = rowbase + l;
        const float* drow = dbl + re * DBLC;
        if (l + 8 < Ll) __builtin_prefetch(dbl + (re + 8) * DBLC, 0, 1);

        const float dtv = dt[re * Ee + e];
        const float xv  = xs[re * Ee + e];
        const float Bn  = drow[Rr + n];
        const float Cn  = drow[Rr + Nst + n];

        const float dA = __expf(dtv * An);
        h = fmaf(dA, h, dtv * xv * Bn);
        float yp = h * Cn;
        yp += __shfl_xor(yp, 1, 32);
        yp += __shfl_xor(yp, 2, 32);
        yp += __shfl_xor(yp, 4, 32);
        yp += __shfl_xor(yp, 8, 32);

        if (n == 0) {
            const float zv  = xz[re * (2 * Ee) + Ee + e];
            const float gat = zv / (1.0f + __expf(-zv));   // silu(z)
            const float yv  = (yp + xv * Dk) * gat;
            y16[re * Ee + e] = (_Float16)yv;
        }
    }
}

// =====================================================================
// Host launch
// =====================================================================
static inline size_t align256(size_t x) { return (x + 255) & ~(size_t)255; }

extern "C" void kernel_launch(void* const* d_in, const int* in_sizes, int n_in,
                              void* d_out, int out_size, void* d_ws, size_t ws_size,
                              hipStream_t stream) {
    const float* hidden = (const float*)d_in[0];   // (B,L,D)
    const float* norm_w = (const float*)d_in[1];   // (D)
    const float* W_in   = (const float*)d_in[2];   // (D, 2E)
    const float* conv_w = (const float*)d_in[3];   // (E, K)
    const float* conv_b = (const float*)d_in[4];   // (E)
    const float* W_x    = (const float*)d_in[5];   // (E, R+2N)
    const float* W_dt   = (const float*)d_in[6];   // (R, E)
    const float* b_dt   = (const float*)d_in[7];   // (E)
    const float* A_log  = (const float*)d_in[8];   // (E, N)
    const float* D_skip = (const float*)d_in[9];   // (E)
    const float* W_out  = (const float*)d_in[10];  // (E, D)
    float* out = (float*)d_out;                    // (B,L,D)

    // ---- workspace layout (all transposed weights are NxK f16) ----
    char* ws = (char*)d_ws;
    size_t off = 0;
    auto alloc = [&](size_t bytes) { char* p = ws + off; off = align256(off + bytes); return p; };
    _Float16* nrm16  = (_Float16*)alloc((size_t)ROWS * Dd * 2);
    _Float16* WinT   = (_Float16*)alloc((size_t)2 * Ee * Dd * 2);   // (2E, D)
    _Float16* WoutT  = (_Float16*)alloc((size_t)Dd * Ee * 2);       // (D, E)
    _Float16* WxT    = (_Float16*)alloc((size_t)DBLC * Ee * 2);     // (64, E)
    _Float16* WdtT   = (_Float16*)alloc((size_t)Ee * Rr * 2);       // (E, R)
    float*    xz     = (float*)   alloc((size_t)ROWS * 2 * Ee * 4);
    float*    xs     = (float*)   alloc((size_t)ROWS * Ee * 4);
    _Float16* xs16   = (_Float16*)alloc((size_t)ROWS * Ee * 2);
    float*    dbl    = (float*)   alloc((size_t)ROWS * DBLC * 4);
    _Float16* dtin16 = (_Float16*)alloc((size_t)ROWS * Rr * 2);
    float*    dt     = (float*)   alloc((size_t)ROWS * Ee * 4);
    _Float16* y16    = (_Float16*)alloc((size_t)ROWS * Ee * 2);
    (void)ws_size;

    // 1) RMSNorm -> f16 activations
    k_rmsnorm<<<ROWS, 128, 0, stream>>>(hidden, norm_w, nrm16);

    // 2) weight transpose + f16 conversion (src is Kdim x Ndim)
    k_cvt_transpose<<<dim3(Dd / 32, 2 * Ee / 32), 256, 0, stream>>>(W_in,  WinT,  Dd, 2 * Ee);
    k_cvt_transpose<<<dim3(Ee / 32, Dd / 32),     256, 0, stream>>>(W_out, WoutT, Ee, Dd);
    k_cvt_transpose<<<dim3(Ee / 32, DBLC / 32),   256, 0, stream>>>(W_x,   WxT,   Ee, DBLC);
    k_cvt_transpose<<<dim3(Rr / 32, Ee / 32),     256, 0, stream>>>(W_dt,  WdtT,  Rr, Ee);

    // 3) xz = nrm @ W_in     (M=8192, K=512, N=2048)
    k_gemm_wmma<<<dim3(2 * Ee / 64, ROWS / 128), 256, 0, stream>>>(
        nrm16, WinT, xz, nullptr, ROWS, 2 * Ee, Dd);

    // 4) depthwise causal conv + SiLU
    k_conv_silu<<<(Bb * Ll * Ee + 255) / 256, 256, 0, stream>>>(xz, conv_w, conv_b, xs, xs16);

    // 5) dbl = xs @ W_x      (M=8192, K=1024, N=64)
    k_gemm_wmma<<<dim3(DBLC / 64, ROWS / 128), 256, 0, stream>>>(
        xs16, WxT, dbl, nullptr, ROWS, DBLC, Ee);

    // 6) dt = softplus(dbl[:, :R] @ W_dt + b_dt)   (M=8192, K=32, N=1024)
    k_dtprep<<<(ROWS * Rr + 255) / 256, 256, 0, stream>>>(dbl, dtin16);
    k_gemm_wmma<<<dim3(Ee / 64, ROWS / 128), 256, 0, stream>>>(
        dtin16, WdtT, dt, nullptr, ROWS, Ee, Rr);
    k_softplus<<<(ROWS * Ee + 255) / 256, 256, 0, stream>>>(dt, b_dt);

    // 7) fused selective scan + skip + silu(z) gate -> y16
    k_scan<<<(Bb * (Ee / 2)) / 8, 256, 0, stream>>>(dt, xs, xz, dbl, A_log, D_skip, y16);

    // 8) out = y @ W_out + hidden   (M=8192, K=1024, N=512, fused residual)
    k_gemm_wmma<<<dim3(Dd / 64, ROWS / 128), 256, 0, stream>>>(
        y16, WoutT, out, hidden, ROWS, Dd, Ee);
}
